// MultiHeadSelfAttention_55293408969108
// MI455X (gfx1250) — compile-verified, matrix operand order checked by reading, not run on hardware
//
#include <hip/hip_runtime.h>
#include <hip/hip_bf16.h>

// ---------------------------------------------------------------------------
// Types for CDNA5 WMMA
// ---------------------------------------------------------------------------
typedef __bf16 bf16_t;
typedef bf16_t bf16x16 __attribute__((ext_vector_type(16)));
typedef float  f32x8   __attribute__((ext_vector_type(8)));
typedef int    i32x4   __attribute__((ext_vector_type(4)));

__device__ __forceinline__ unsigned short f32_to_bf16_bits(float f) {
    union { float f; unsigned int u; } c; c.f = f;
    unsigned int u = c.u;
    unsigned int r = u + 0x7FFFu + ((u >> 16) & 1u);   // round-to-nearest-even
    return (unsigned short)(r >> 16);
}
__device__ __forceinline__ float bf16_bits_to_f32(unsigned short h) {
    union { unsigned int u; float f; } c; c.u = ((unsigned int)h) << 16;
    return c.f;
}

union FragU { bf16x16 v; uint4 q[2]; };

// ---------------------------------------------------------------------------
// Async global->LDS (gfx1250 GLOBAL_LOAD_ASYNC_TO_LDS_B128, ASYNCcnt) with
// a safe fallback if the builtin is not available in this toolchain.
// Builtin signature (from probe): (i32x4 as1*, i32x4 as3*, Ii offset, Ii cpol)
// ---------------------------------------------------------------------------
#if defined(__HIP_DEVICE_COMPILE__) && defined(__has_builtin)
#if __has_builtin(__builtin_amdgcn_global_load_async_to_lds_b128)
#define HAVE_ASYNC_LDS 1
#endif
#endif

#if HAVE_ASYNC_LDS
typedef __attribute__((address_space(1))) i32x4 gbl_i32x4;
typedef __attribute__((address_space(3))) i32x4 lds_i32x4;
#endif

__device__ __forceinline__ void copy16_to_lds(const void* g, void* l) {
#if HAVE_ASYNC_LDS
    __builtin_amdgcn_global_load_async_to_lds_b128(
        (gbl_i32x4*)(unsigned long long)g,
        (lds_i32x4*)(unsigned int)(unsigned long long)l,
        0, 0);
#else
    *(uint4*)l = *(const uint4*)g;
#endif
}
__device__ __forceinline__ void lds_copy_wait() {
#if HAVE_ASYNC_LDS
#if __has_builtin(__builtin_amdgcn_s_wait_asynccnt)
    __builtin_amdgcn_s_wait_asynccnt(0);
#else
    asm volatile("s_wait_asynccnt 0x0" ::: "memory");
#endif
#endif
}

// ---------------------------------------------------------------------------
// Kernel 1: fp32 -> bf16 conversion (vectorized: float4 in, uint2 out)
// ---------------------------------------------------------------------------
__global__ void cvt_f32_bf16(const float4* __restrict__ src,
                             uint2* __restrict__ dst, int n4) {
    int i = blockIdx.x * blockDim.x + threadIdx.x;
    if (i >= n4) return;
    float4 f = src[i];
    unsigned int lo = (unsigned int)f32_to_bf16_bits(f.x) |
                      ((unsigned int)f32_to_bf16_bits(f.y) << 16);
    unsigned int hi = (unsigned int)f32_to_bf16_bits(f.z) |
                      ((unsigned int)f32_to_bf16_bits(f.w) << 16);
    dst[i] = make_uint2(lo, hi);
}

// ---------------------------------------------------------------------------
// Kernel 2: bf16 GEMM  C[M,N] = A[M,K] * B[N,K]^T  (both row-major over K)
// Block tile 128(M) x 128(N), K-step 32, double-buffered async LDS staging.
// 256 threads = 8 waves arranged 4(M) x 2(N); wave tile 32x64 = 2x4 WMMA
// fragments -> 8 v_wmma per K-step per wave from 12 ds_load_b128.
// mode 0: store fp32 to C [M,N]
// mode 1: store bf16 to qkv layout [B=2,H=16,S=2048,dk=64] (N=1024 cols)
// ---------------------------------------------------------------------------
#define GBM 128
#define GBN 128
#define GBK 32
#define GLD 40   // LDS row stride in ushort (80 B, 16B-aligned)

__global__ void __launch_bounds__(256)
gemm_bf16(const unsigned short* __restrict__ A,
          const unsigned short* __restrict__ Bm,
          void* __restrict__ Cout,
          int M, int N, int K, int mode)
{
    __shared__ unsigned short As[2][GBM * GLD];
    __shared__ unsigned short Bs[2][GBN * GLD];

    const int tid  = threadIdx.x;
    const int wave = tid >> 5;
    const int lane = tid & 31;
    const int l16  = lane & 15;
    const int lh   = lane >> 4;        // lane-half: K sub-select
    const int wm   = wave & 3;         // 0..3 -> 32-row strip
    const int wn   = wave >> 2;        // 0..1 -> 64-col strip
    const int blockM = blockIdx.y * GBM;
    const int blockN = blockIdx.x * GBN;

    f32x8 acc[2][4];
#pragma unroll
    for (int i = 0; i < 2; ++i)
#pragma unroll
        for (int j = 0; j < 4; ++j)
            acc[i][j] = f32x8{0.f,0.f,0.f,0.f,0.f,0.f,0.f,0.f};

    // stage one 128x32 A tile + 128x32 B tile into buffer `buf`
    auto stage = [&](int buf, int kk) {
#pragma unroll
        for (int c = 0; c < 2; ++c) {
            int lin = tid * 16 + c * 8;
            int row = lin >> 5;
            int col = lin & 31;
            copy16_to_lds(A + (size_t)(blockM + row) * K + kk + col,
                          &As[buf][row * GLD + col]);
            copy16_to_lds(Bm + (size_t)(blockN + row) * K + kk + col,
                          &Bs[buf][row * GLD + col]);
        }
    };

    int cur = 0;
    stage(0, 0);
    lds_copy_wait();
    __syncthreads();

    for (int k0 = 0; k0 < K; k0 += GBK) {
        int nxt = cur ^ 1;
        if (k0 + GBK < K) stage(nxt, k0 + GBK);   // overlap with compute

        // ---- fragment loads from current buffer ----
        bf16x16 aF[2], bF[4];
#pragma unroll
        for (int i = 0; i < 2; ++i) {
            const unsigned short* p = &As[cur][(wm * 32 + i * 16 + l16) * GLD + lh * 8];
            FragU t;
            t.q[0] = *(const uint4*)(p);
            t.q[1] = *(const uint4*)(p + 16);
            aF[i] = t.v;
        }
#pragma unroll
        for (int j = 0; j < 4; ++j) {
            const unsigned short* p = &Bs[cur][(wn * 64 + j * 16 + l16) * GLD + lh * 8];
            FragU t;
            t.q[0] = *(const uint4*)(p);
            t.q[1] = *(const uint4*)(p + 16);
            bF[j] = t.v;
        }
        // ---- 8 WMMAs ----
#pragma unroll
        for (int i = 0; i < 2; ++i)
#pragma unroll
            for (int j = 0; j < 4; ++j)
                acc[i][j] = __builtin_amdgcn_wmma_f32_16x16x32_bf16(
                    false, aF[i], false, bF[j], (short)0, acc[i][j], false, false);

        lds_copy_wait();     // next buffer fully written by this wave
        __syncthreads();     // .. and by all other waves; reads of cur done
        cur = nxt;
    }

    // ---- store: C-layout lane=(col l16), rows r + 8*lh ----
#pragma unroll
    for (int i = 0; i < 2; ++i)
#pragma unroll
        for (int j = 0; j < 4; ++j)
#pragma unroll
            for (int r = 0; r < 8; ++r) {
                int m = blockM + wm * 32 + i * 16 + r + lh * 8;
                int n = blockN + wn * 64 + j * 16 + l16;
                if (mode == 0) {
                    ((float*)Cout)[(size_t)m * N + n] = acc[i][j][r];
                } else {
                    int b = m >> 11, s = m & 2047;
                    int h = n >> 6,  d = n & 63;
                    ((unsigned short*)Cout)[(((size_t)b * 16 + h) * 2048 + s) * 64 + d] =
                        f32_to_bf16_bits(acc[i][j][r]);
                }
            }
}

// ---------------------------------------------------------------------------
// Kernel 3: RoPE in-place on q,k  [B=2,H=16,S=2048,dk=64] bf16
// ---------------------------------------------------------------------------
__global__ void rope_kernel(unsigned short* __restrict__ q,
                            unsigned short* __restrict__ k,
                            const int* __restrict__ tp)
{
    int idx = blockIdx.x * blockDim.x + threadIdx.x;
    if (idx >= 2 * 16 * 2048 * 32) return;
    int p  = idx & 31;
    int s  = (idx >> 5) & 2047;
    int bh = idx >> 16;          // 0..31
    int b  = bh >> 4;

    float pos  = (float)tp[b * 2048 + s];
    float freq = __expf(-((float)(2 * p) * (1.0f / 64.0f)) * 9.2103403719761836f);
    float sn, cs;
    __sincosf(pos * freq, &sn, &cs);

    size_t base = ((size_t)bh * 2048 + s) * 64 + 2 * p;
    {
        float x1 = bf16_bits_to_f32(q[base]);
        float x2 = bf16_bits_to_f32(q[base + 1]);
        q[base]     = f32_to_bf16_bits(x1 * cs - x2 * sn);
        q[base + 1] = f32_to_bf16_bits(x1 * sn + x2 * cs);
    }
    {
        float x1 = bf16_bits_to_f32(k[base]);
        float x2 = bf16_bits_to_f32(k[base + 1]);
        k[base]     = f32_to_bf16_bits(x1 * cs - x2 * sn);
        k[base + 1] = f32_to_bf16_bits(x1 * sn + x2 * cs);
    }
}

// ---------------------------------------------------------------------------
// Kernel 4: causal flash attention, per (b,h,qtile-of-64).
// 128 threads = 4 waves; wave handles a 16-row q strip. kv tiles of 64.
// ---------------------------------------------------------------------------
#define VST 72   // LDS row stride ushort (144 B)

__global__ void __launch_bounds__(128)
attn_kernel(const unsigned short* __restrict__ Q,
            const unsigned short* __restrict__ Kc,
            const unsigned short* __restrict__ V,
            unsigned short* __restrict__ O)
{
    __shared__ unsigned short Ps[4][16 * VST];  // per-wave P scratch 16x64
    __shared__ unsigned short Vt[64 * VST];     // V^T tile: [dk][kv]

    const int S = 2048, DK = 64, H = 16;
    const int tid  = threadIdx.x;
    const int wave = tid >> 5;
    const int lane = tid & 31;
    const int l16  = lane & 15;
    const int lh   = lane >> 4;
    const int qt   = blockIdx.x;  // 0..31
    const int h    = blockIdx.y;  // 0..15
    const int b    = blockIdx.z;  // 0..1

    const size_t head = ((size_t)b * H + h) * S * DK;
    const unsigned short* Qh = Q  + head;
    const unsigned short* Kh = Kc + head;
    const unsigned short* Vh = V  + head;
    const int qbase = qt * 64 + wave * 16;

    // ---- Q fragments (A-layout), resident in registers ----
    bf16x16 qF[2];
#pragma unroll
    for (int c = 0; c < 2; ++c) {
        const unsigned short* p = Qh + (size_t)(qbase + l16) * DK + c * 32 + lh * 8;
        FragU t;
        t.q[0] = *(const uint4*)(p);
        t.q[1] = *(const uint4*)(p + 16);
        qF[c] = t.v;
    }

    f32x8 oAcc[4];
    float mI[8], lI[8];
#pragma unroll
    for (int nt = 0; nt < 4; ++nt)
        oAcc[nt] = f32x8{0.f,0.f,0.f,0.f,0.f,0.f,0.f,0.f};
#pragma unroll
    for (int r = 0; r < 8; ++r) { mI[r] = -3.0e38f; lI[r] = 0.f; }

    const float scale = 0.125f;  // 1/sqrt(64)
    const int nkt = qt + 1;      // causal: kv tiles 0..qt

    for (int kt = 0; kt < nkt; ++kt) {
        const int kvbase = kt * 64;
        __syncthreads();  // protect Vt from previous iteration readers

        // ---- stage V^T into LDS (transpose during scatter) ----
#pragma unroll
        for (int c = 0; c < 4; ++c) {
            int lin = tid * 8 + c * 1024;
            int row = lin >> 6;   // kv within tile
            int col = lin & 63;   // dk
            uint4 d4 = *(const uint4*)(Vh + (size_t)(kvbase + row) * DK + col);
            const unsigned short* e = (const unsigned short*)&d4;
#pragma unroll
            for (int i = 0; i < 8; ++i) Vt[(col + i) * VST + row] = e[i];
        }
        if (kt + 1 < nkt)
            __builtin_prefetch(Vh + (size_t)(kvbase + 64 + (tid >> 1)) * DK, 0, 0);
        __syncthreads();

        // ---- scores S = (Q K^T) * scale, K frags direct from global ----
        f32x8 sF[4];
#pragma unroll
        for (int nt = 0; nt < 4; ++nt) {
            f32x8 a = f32x8{0.f,0.f,0.f,0.f,0.f,0.f,0.f,0.f};
#pragma unroll
            for (int c = 0; c < 2; ++c) {
                const unsigned short* p =
                    Kh + (size_t)(kvbase + nt * 16 + l16) * DK + c * 32 + lh * 8;
                FragU t;
                t.q[0] = *(const uint4*)(p);
                t.q[1] = *(const uint4*)(p + 16);
                a = __builtin_amdgcn_wmma_f32_16x16x32_bf16(
                        false, qF[c], false, t.v, (short)0, a, false, false);
            }
#pragma unroll
            for (int r = 0; r < 8; ++r) a[r] *= scale;
            sF[nt] = a;
        }

        // ---- causal mask on diagonal tile ----
        if (kt == qt) {
#pragma unroll
            for (int nt = 0; nt < 4; ++nt)
#pragma unroll
                for (int r = 0; r < 8; ++r) {
                    int kj = kvbase + nt * 16 + l16;
                    int qi = qbase + r + lh * 8;
                    if (kj > qi) sF[nt][r] = -1.0e9f;
                }
        }

        // ---- online softmax: row max over 64 cols (4 frags x 16 lanes) ----
        float rm[8];
#pragma unroll
        for (int r = 0; r < 8; ++r) {
            float m = sF[0][r];
#pragma unroll
            for (int nt = 1; nt < 4; ++nt) m = fmaxf(m, sF[nt][r]);
#pragma unroll
            for (int off = 1; off < 16; off <<= 1)
                m = fmaxf(m, __shfl_xor(m, off, 16));
            rm[r] = m;
        }

        float mNew[8], corr[8], rs[8];
#pragma unroll
        for (int r = 0; r < 8; ++r) {
            mNew[r] = fmaxf(mI[r], rm[r]);
            corr[r] = __expf(mI[r] - mNew[r]);
            mI[r]   = mNew[r];
            rs[r]   = 0.f;
        }

        f32x8 pF[4];
#pragma unroll
        for (int nt = 0; nt < 4; ++nt)
#pragma unroll
            for (int r = 0; r < 8; ++r) {
                float pv = __expf(sF[nt][r] - mNew[r]);
                pF[nt][r] = pv;
                rs[r] += pv;
            }
#pragma unroll
        for (int r = 0; r < 8; ++r) {
#pragma unroll
            for (int off = 1; off < 16; off <<= 1)
                rs[r] += __shfl_xor(rs[r], off, 16);
            lI[r] = lI[r] * corr[r] + rs[r];
        }
#pragma unroll
        for (int nt = 0; nt < 4; ++nt)
#pragma unroll
            for (int r = 0; r < 8; ++r) oAcc[nt][r] *= corr[r];

        // ---- P: C-layout -> LDS -> A-layout (bf16) ----
#pragma unroll
        for (int nt = 0; nt < 4; ++nt)
#pragma unroll
            for (int r = 0; r < 8; ++r)
                Ps[wave][(r + lh * 8) * VST + nt * 16 + l16] =
                    f32_to_bf16_bits(pF[nt][r]);
        __syncthreads();

        bf16x16 pA[2];
#pragma unroll
        for (int c = 0; c < 2; ++c) {
            const unsigned short* p = &Ps[wave][l16 * VST + c * 32 + lh * 8];
            FragU t;
            t.q[0] = *(const uint4*)(p);
            t.q[1] = *(const uint4*)(p + 16);
            pA[c] = t.v;
        }

        // ---- O += P @ V ; V B-frags from transposed LDS (contiguous) ----
#pragma unroll
        for (int nt = 0; nt < 4; ++nt) {
#pragma unroll
            for (int c = 0; c < 2; ++c) {
                const unsigned short* p =
                    &Vt[(nt * 16 + l16) * VST + c * 32 + lh * 8];
                FragU t;
                t.q[0] = *(const uint4*)(p);
                t.q[1] = *(const uint4*)(p + 16);
                oAcc[nt] = __builtin_amdgcn_wmma_f32_16x16x32_bf16(
                               false, pA[c], false, t.v, (short)0, oAcc[nt], false, false);
            }
        }
    }

    // ---- epilogue: O /= l, store merged-head bf16 [B,S,1024] ----
#pragma unroll
    for (int r = 0; r < 8; ++r) {
        float inv = (lI[r] > 0.f) ? (1.0f / lI[r]) : 0.f;
#pragma unroll
        for (int nt = 0; nt < 4; ++nt) {
            int qi = qbase + r + lh * 8;
            int n  = nt * 16 + l16;
            O[((size_t)b * 2048 + qi) * 1024 + h * 64 + n] =
                f32_to_bf16_bits(oAcc[nt][r] * inv);
        }
    }
}

// ---------------------------------------------------------------------------
// Host launcher
// ---------------------------------------------------------------------------
extern "C" void kernel_launch(void* const* d_in, const int* in_sizes, int n_in,
                              void* d_out, int out_size, void* d_ws, size_t ws_size,
                              hipStream_t stream) {
    (void)in_sizes; (void)n_in; (void)out_size; (void)ws_size;

    const float* x  = (const float*)d_in[0];
    const int*   tp = (const int*)d_in[1];
    const float* Wq = (const float*)d_in[2];
    const float* Wk = (const float*)d_in[3];
    const float* Wv = (const float*)d_in[4];
    const float* Wo = (const float*)d_in[5];
    float* out = (float*)d_out;

    const size_t M = 4096;        // B*S
    const size_t D = 1024;
    const size_t QKV = 2ull * 16 * 2048 * 64;  // = 4 Mi elements

    char* w = (char*)d_ws;
    unsigned short* xb  = (unsigned short*)w;  w += M * D * 2;
    unsigned short* Wqb = (unsigned short*)w;  w += D * D * 2;
    unsigned short* Wkb = (unsigned short*)w;  w += D * D * 2;
    unsigned short* Wvb = (unsigned short*)w;  w += D * D * 2;
    unsigned short* Wob = (unsigned short*)w;  w += D * D * 2;
    unsigned short* qA  = (unsigned short*)w;  w += QKV * 2;
    unsigned short* kA  = (unsigned short*)w;  w += QKV * 2;
    unsigned short* vA  = (unsigned short*)w;  w += QKV * 2;
    unsigned short* Ob  = (unsigned short*)w;  w += M * D * 2;

    // 1) convert to bf16 (vectorized x4)
    {
        int n4 = (int)(M * D / 4);
        cvt_f32_bf16<<<(n4 + 255) / 256, 256, 0, stream>>>((const float4*)x,  (uint2*)xb,  n4);
        n4 = (int)(D * D / 4);
        cvt_f32_bf16<<<(n4 + 255) / 256, 256, 0, stream>>>((const float4*)Wq, (uint2*)Wqb, n4);
        cvt_f32_bf16<<<(n4 + 255) / 256, 256, 0, stream>>>((const float4*)Wk, (uint2*)Wkb, n4);
        cvt_f32_bf16<<<(n4 + 255) / 256, 256, 0, stream>>>((const float4*)Wv, (uint2*)Wvb, n4);
        cvt_f32_bf16<<<(n4 + 255) / 256, 256, 0, stream>>>((const float4*)Wo, (uint2*)Wob, n4);
    }

    // 2) QKV projections (WMMA GEMM, head-split bf16 store)
    dim3 ggrid(D / GBN, M / GBM);   // (8, 32)
    gemm_bf16<<<ggrid, 256, 0, stream>>>(xb, Wqb, qA, (int)M, (int)D, (int)D, 1);
    gemm_bf16<<<ggrid, 256, 0, stream>>>(xb, Wkb, kA, (int)M, (int)D, (int)D, 1);
    gemm_bf16<<<ggrid, 256, 0, stream>>>(xb, Wvb, vA, (int)M, (int)D, (int)D, 1);

    // 3) RoPE on q,k
    {
        int n = 2 * 16 * 2048 * 32;
        rope_kernel<<<(n + 255) / 256, 256, 0, stream>>>(qA, kA, tp);
    }

    // 4) causal flash attention
    attn_kernel<<<dim3(32, 16, 2), 128, 0, stream>>>(qA, kA, vA, Ob);

    // 5) output projection -> fp32 d_out
    gemm_bf16<<<ggrid, 256, 0, stream>>>(Ob, Wob, out, (int)M, (int)D, (int)D, 0);
}
